// TripletHard_26482768347623
// MI455X (gfx1250) — matrix operand order repaired and checked by reading, not compiled, not run on hardware
//
#include <hip/hip_runtime.h>

typedef _Float16 half_t;
typedef __attribute__((ext_vector_type(16))) _Float16 v16h;
typedef __attribute__((ext_vector_type(8)))  _Float16 v8h;
typedef __attribute__((ext_vector_type(4)))  _Float16 v4h;
typedef __attribute__((ext_vector_type(8)))  float    v8f;

#define B_N       8192
#define D_K       128
#define MARGIN_F  1.0f
#define EPS_F     1e-5f
#define CHUNKS    4
#define COLS_PER_CHUNK (B_N / CHUNKS)      // 2048
#define COL_TILES (COLS_PER_CHUNK / 16)    // 128
#define ROW_TILES (B_N / 16)               // 512
#define WAVES_PER_BLOCK 8
#define MAIN_BLOCKS (ROW_TILES / WAVES_PER_BLOCK)  // 64

static __device__ __forceinline__ v16h pack16(v8h lo, v8h hi) {
  return __builtin_shufflevector(lo, hi, 0,1,2,3,4,5,6,7,8,9,10,11,12,13,14,15);
}

// ---------------------------------------------------------------------------
// Kernel 1: one wave per row. Convert f32 -> f16 and compute exact fp32 row
// squared-norm (so only the -2*G cross term is f16-sourced).
// ---------------------------------------------------------------------------
__global__ __launch_bounds__(256) void prep_kernel(const float* __restrict__ f,
                                                   half_t* __restrict__ fh,
                                                   float* __restrict__ sq) {
  const int row  = (int)((blockIdx.x * blockDim.x + threadIdx.x) >> 5);
  const int lane = (int)(threadIdx.x & 31);
  if (row >= B_N) return;
  const float4 v = ((const float4*)(f + (size_t)row * D_K))[lane];  // 4 elems/lane
  float s = v.x * v.x + v.y * v.y + v.z * v.z + v.w * v.w;
  v4h h4 = { (half_t)v.x, (half_t)v.y, (half_t)v.z, (half_t)v.w };
  *(v4h*)(fh + (size_t)row * D_K + lane * 4) = h4;
#pragma unroll
  for (int m = 16; m >= 1; m >>= 1) s += __shfl_xor(s, m, 32);
  if (lane == 0) sq[row] = s;
}

// ---------------------------------------------------------------------------
// Kernel 2: fused GEMM + selection. Each wave owns a 16-row tile and streams a
// 2048-column chunk in 16-column steps. Per step: 4x v_wmma_f32_16x16x32_f16
// chained over K=128, then squared distances d2 = sq[m]+sq[n]-2G+eps tracked
// through online (min1,min2)-same / min-diff per row. Cross-lane butterfly
// reduction at the end; per-(row,chunk) triples written to workspace.
// ---------------------------------------------------------------------------
__global__ __launch_bounds__(256) void tile_kernel(const half_t* __restrict__ fh,
                                                   const float* __restrict__ sq,
                                                   const int* __restrict__ lab,
                                                   float* __restrict__ m1out,
                                                   float* __restrict__ m2out,
                                                   float* __restrict__ mdout) {
  const int lane = (int)(threadIdx.x & 31);
  const int wave = (int)(threadIdx.x >> 5);
  const int rowTile = (int)blockIdx.x * WAVES_PER_BLOCK + wave;  // 0..511
  const int chunk   = (int)blockIdx.y;                           // 0..3
  const int rb = rowTile * 16;
  const int colBase = chunk * COLS_PER_CHUNK;
  const int h  = lane >> 4;   // half of wave (0/1)
  const int mr = lane & 15;   // row index within A tile / col index within B tile
  const float FINF = __builtin_inff();

  // --- Preload A fragments for K = 0..127 (4 ksteps). 16-bit A 16x32 layout:
  // lane half h, VGPR v<4 -> K = 8h+2v..; v>=4 -> K = 16+8h+... : two
  // contiguous 16-byte chunks per lane at k-offsets 8h and 16+8h.
  const half_t* arow = fh + (size_t)(rb + mr) * D_K;
  v16h afrag[4];
#pragma unroll
  for (int t = 0; t < 4; ++t) {
    v8h lo = *(const v8h*)(arow + 32 * t + 8 * h);
    v8h hi = *(const v8h*)(arow + 32 * t + 16 + 8 * h);
    afrag[t] = pack16(lo, hi);
  }

  // C/D VGPR r holds (M = r + 8h, N = lane&15); preload per-row norm + label.
  float sqm[8]; int labm[8];
#pragma unroll
  for (int r = 0; r < 8; ++r) {
    sqm[r]  = sq[rb + 8 * h + r] + EPS_F;
    labm[r] = lab[rb + 8 * h + r];
  }

  float min1[8], min2[8], mind[8];
#pragma unroll
  for (int r = 0; r < 8; ++r) { min1[r] = FINF; min2[r] = FINF; mind[r] = FINF; }

  for (int ct = 0; ct < COL_TILES; ++ct) {
    const int c0 = colBase + ct * 16;
    // 16-bit B 32x16 layout (same k order as A): two contiguous 16-byte
    // chunks per lane at k-offsets 8h and 16+8h.
    const half_t* brow = fh + (size_t)(c0 + mr) * D_K;
    v8f acc = {0.f, 0.f, 0.f, 0.f, 0.f, 0.f, 0.f, 0.f};
#pragma unroll
    for (int t = 0; t < 4; ++t) {
      const v8h* bp = (const v8h*)(brow + 32 * t + 8 * h);
      v16h bfrag = pack16(bp[0], bp[2]);
      acc = __builtin_amdgcn_wmma_f32_16x16x32_f16(
          /*neg_a=*/false, afrag[t], /*neg_b=*/false, bfrag,
          /*c_mod=*/(short)0, acc, /*reuse_a=*/false, /*reuse_b=*/false);
    }
    const float sqn  = sq[c0 + mr];
    const int   labn = lab[c0 + mr];
#pragma unroll
    for (int r = 0; r < 8; ++r) {
      float d2 = fmaf(acc[r], -2.0f, sqm[r] + sqn);
      d2 = fmaxf(d2, 0.0f);
      const bool  same = (labm[r] == labn);
      const float ds = same ? d2 : FINF;
      const float dd = same ? FINF : d2;
      mind[r] = fminf(mind[r], dd);
      const float mx = fmaxf(min1[r], ds);   // online 2-smallest update
      min1[r] = fminf(min1[r], ds);
      min2[r] = fminf(min2[r], mx);
    }
  }

  // Butterfly reduction across the 16 lanes of each wave half (N dimension).
#pragma unroll
  for (int r = 0; r < 8; ++r) {
#pragma unroll
    for (int m = 1; m < 16; m <<= 1) {
      const float o1 = __shfl_xor(min1[r], m, 32);
      const float o2 = __shfl_xor(min2[r], m, 32);
      const float od = __shfl_xor(mind[r], m, 32);
      mind[r] = fminf(mind[r], od);
      const float mx = fmaxf(min1[r], o1);   // merge sorted pairs
      min2[r] = fminf(fminf(min2[r], o2), mx);
      min1[r] = fminf(min1[r], o1);
    }
  }

  if ((lane & 15) == 0) {   // lane 0 -> rows rb..rb+7, lane 16 -> rb+8..rb+15
    const int rowBase = rb + 8 * h;
    const size_t coff = (size_t)chunk * B_N;
#pragma unroll
    for (int r = 0; r < 8; ++r) {
      m1out[coff + rowBase + r] = min1[r];
      m2out[coff + rowBase + r] = min2[r];
      mdout[coff + rowBase + r] = mind[r];
    }
  }
}

// ---------------------------------------------------------------------------
// Kernel 3: merge the CHUNKS partial triples per row, take sqrt, hinge, and
// block-reduce (deterministic, no float atomics). 32 blocks x 256 threads.
// ---------------------------------------------------------------------------
__global__ __launch_bounds__(256) void merge_kernel(const float* __restrict__ m1a,
                                                    const float* __restrict__ m2a,
                                                    const float* __restrict__ mda,
                                                    float* __restrict__ partial) {
  const int row = (int)(blockIdx.x * blockDim.x + threadIdx.x);
  const float FINF = __builtin_inff();
  float m1 = FINF, m2 = FINF, md = FINF;
#pragma unroll
  for (int c = 0; c < CHUNKS; ++c) {
    const float a1 = m1a[(size_t)c * B_N + row];
    const float a2 = m2a[(size_t)c * B_N + row];
    const float ad = mda[(size_t)c * B_N + row];
    const float n1 = fminf(m1, a1);
    const float n2 = fminf(fmaxf(m1, a1), fminf(m2, a2));
    m1 = n1; m2 = n2;
    md = fminf(md, ad);
  }
  float hinge = fmaxf(MARGIN_F + sqrtf(m2) - sqrtf(md), 0.0f);
#pragma unroll
  for (int m = 16; m >= 1; m >>= 1) hinge += __shfl_xor(hinge, m, 32);
  __shared__ float wsum[8];
  const int wave = (int)(threadIdx.x >> 5);
  if ((threadIdx.x & 31) == 0) wsum[wave] = hinge;
  __syncthreads();
  if (threadIdx.x == 0) {
    float s = 0.f;
#pragma unroll
    for (int w = 0; w < 8; ++w) s += wsum[w];
    partial[blockIdx.x] = s;
  }
}

// ---------------------------------------------------------------------------
// Kernel 4: final scalar reduction of the 32 block partials, scale by 1/B.
// ---------------------------------------------------------------------------
__global__ void final_kernel(const float* __restrict__ partial, float* __restrict__ out) {
  const int lane = (int)threadIdx.x;  // 32 threads
  float s = partial[lane];
#pragma unroll
  for (int m = 16; m >= 1; m >>= 1) s += __shfl_xor(s, m, 32);
  if (lane == 0) out[0] = s * (1.0f / (float)B_N);
}

// ---------------------------------------------------------------------------
extern "C" void kernel_launch(void* const* d_in, const int* in_sizes, int n_in,
                              void* d_out, int out_size, void* d_ws, size_t ws_size,
                              hipStream_t stream) {
  (void)in_sizes; (void)n_in; (void)out_size; (void)ws_size;
  const float* feature = (const float*)d_in[0];
  const int*   label   = (const int*)d_in[1];
  float* out = (float*)d_out;

  char* ws = (char*)d_ws;
  half_t* fh = (half_t*)ws;                                   // 2 MB
  float*  sq = (float*)(ws + (1u << 21));                     // 32 KB
  float*  m1 = (float*)(ws + (1u << 21) + (1u << 15));        // 128 KB
  float*  m2 = m1 + (size_t)CHUNKS * B_N;                     // 128 KB
  float*  md = m2 + (size_t)CHUNKS * B_N;                     // 128 KB
  float*  pt = md + (size_t)CHUNKS * B_N;                     // 32 floats

  prep_kernel<<<B_N / 8, 256, 0, stream>>>(feature, fh, sq);
  tile_kernel<<<dim3(MAIN_BLOCKS, CHUNKS), 256, 0, stream>>>(fh, sq, label, m1, m2, md);
  merge_kernel<<<B_N / 256, 256, 0, stream>>>(m1, m2, md, pt);
  final_kernel<<<1, 32, 0, stream>>>(pt, out);
}
